// ConvLinformer_70411693851103
// MI455X (gfx1250) — matrix operand order repaired and checked by reading, not compiled, not run on hardware
//
#include <hip/hip_runtime.h>
#include <hip/hip_bf16.h>
#include <cstdint>
#include <cstddef>

// ============================================================================
// ConvLinformer forward for MI455X (gfx1250, wave32, WMMA).
// All matmuls run through v_wmma_f32_16x16x32_bf16 (bf16 in, f32 accumulate).
// Tile staging uses CDNA5 async global->LDS copies (ASYNCcnt) where layout
// permits; B-natural tiles are transposed during the LDS store.
// ============================================================================

// ---- model dims (match reference) ----
constexpr int cB = 2, cN = 4096, cD = 1024, cH = 8, cDH = 128;
constexpr int cK = 256, cS = 16, cDFF = 4096, cL = 2;

typedef unsigned short u16t;
typedef __attribute__((ext_vector_type(16))) __bf16 v16bf;
typedef __attribute__((ext_vector_type(8)))  float  v8f;

union Frag   { v16bf v; uint4 q[2]; };
union AccU   { v8f v; float f[8]; };

__device__ __forceinline__ u16t f2bf(float f) {
  union { float f; unsigned u; } cv; cv.f = f;
  unsigned u = cv.u + 0x7FFFu + ((cv.u >> 16) & 1u);   // round-to-nearest-even
  return (u16t)(u >> 16);
}
__device__ __forceinline__ float gelu_exact(float x) {
  return 0.5f * x * (1.0f + erff(x * 0.7071067811865475f));
}

// async 16-byte global -> LDS copy (per-lane), tracked by ASYNCcnt
__device__ __forceinline__ void async_cp16(const void* gptr, void* lptr) {
  unsigned l = (unsigned)(size_t)lptr;    // low 32 bits of shared ptr = ds addr
  asm volatile("global_load_async_to_lds_b128 %0, %1, off"
               :: "v"(l), "v"(gptr) : "memory");
}
__device__ __forceinline__ void async_wait0() {
  asm volatile("s_wait_asynccnt 0x0" ::: "memory");
}

// ---------------------------------------------------------------------------
// WMMA GEMM: C[M,N] = alpha * A[M,K]·B[K,N] (+bias) (+gelu), A/B bf16, C f32/bf16
//   BT=true : Bm points to B^T stored [N,K] row-major (ldb = K-dir stride)
//   BT=false: Bm points to B stored [K,N] row-major (transposed into LDS)
// Block tile 128x64, BK=64 (2 wmma k-steps/stage), 8 wave32 waves.
// ---------------------------------------------------------------------------
struct GP {
  const u16t* A; const u16t* Bm; void* C; const float* bias;
  int M, N, K, lda, ldb, ldc, HD;
  long long sA1, sA2, sB1, sB2, sC1, sC2;
  float alpha;
};

constexpr int BM = 128, BN = 64, BK = 64, LDP = 72;  // 72*2B=144B row: 16B aligned + bank shift

template<bool BT, int EPI, bool OBF>
__global__ __launch_bounds__(256) void k_gemm(GP p) {
  __shared__ __align__(16) u16t sA[2][BM][LDP];
  __shared__ __align__(16) u16t sB[2][BN][LDP];

  const int tid  = threadIdx.x;
  const int lane = tid & 31, wave = tid >> 5;
  const int wm = wave & 3, wn = wave >> 2;       // 4x2 wave grid
  const int l16 = lane & 15, hi = lane >> 4;

  const int z  = blockIdx.z;
  const long long z1 = z / p.HD, z2 = z % p.HD;
  const u16t* gA = p.A  + z1 * p.sA1 + z2 * p.sA2;
  const u16t* gB = p.Bm + z1 * p.sB1 + z2 * p.sB2;
  const long long coff = z1 * p.sC1 + z2 * p.sC2;

  const int rowBase = blockIdx.y * BM;
  const int colBase = blockIdx.x * BN;

  // --- async stage: A tile (128x64 = 4 chunks/thread), B^T tile (2 chunks) ---
  auto issueAsync = [&](int kt, int buf) {
    const long long kB = (long long)kt * BK;
#pragma unroll
    for (int i = 0; i < 4; ++i) {
      int c = tid + i * 256;
      int r = c >> 3, kc = (c & 7) << 3;
      async_cp16(gA + (long long)(rowBase + r) * p.lda + kB + kc, &sA[buf][r][kc]);
    }
    if (BT) {
#pragma unroll
      for (int i = 0; i < 2; ++i) {
        int c = tid + i * 256;
        int r = c >> 3, kc = (c & 7) << 3;
        async_cp16(gB + (long long)(colBase + r) * p.ldb + kB + kc, &sB[buf][r][kc]);
      }
    }
  };
  // --- B natural layout: regs + transpose during LDS store ---
  uint4 rb[2];
  auto loadBnat = [&](int kt) {
    if (!BT) {
      const long long kB = (long long)kt * BK;
#pragma unroll
      for (int i = 0; i < 2; ++i) {
        int c = tid + i * 256;
        int kr = c >> 3, nc = (c & 7) << 3;
        rb[i] = *(const uint4*)(gB + (kB + kr) * (long long)p.ldb + colBase + nc);
      }
    }
  };
  auto storeBnat = [&](int buf) {
    if (!BT) {
#pragma unroll
      for (int i = 0; i < 2; ++i) {
        int c = tid + i * 256;
        int kr = c >> 3, nc = (c & 7) << 3;
        const u16t* pv = (const u16t*)&rb[i];
#pragma unroll
        for (int j = 0; j < 8; ++j) sB[buf][nc + j][kr] = pv[j];
      }
    }
  };

  v8f acc[2][2] = {};
  const int nk = p.K / BK;

  issueAsync(0, 0); loadBnat(0);
  async_wait0(); storeBnat(0);
  __syncthreads();

  for (int kt = 0; kt < nk; ++kt) {
    const int cur = kt & 1;
    const bool more = (kt + 1 < nk);
    if (more) { issueAsync(kt + 1, cur ^ 1); loadBnat(kt + 1); }

#pragma unroll
    for (int ks = 0; ks < 2; ++ks) {             // two 16x16x32 k-steps per stage
      Frag af[2], bfr[2];
#pragma unroll
      for (int f = 0; f < 2; ++f) {
        // A 16x32 frag: lane row M=l16; e0..7 -> K=hi*8.., e8..15 -> K=hi*8+16..
        const u16t* pa = &sA[cur][wm * 32 + f * 16 + l16][ks * 32 + hi * 8];
        af[f].q[0] = *(const uint4*)pa;
        af[f].q[1] = *(const uint4*)(pa + 16);
        // B 32x16 frag: lane col N=l16; e0..7 -> K=hi*8.., e8..15 -> K=hi*8+16.. (B^T rows)
        const u16t* pb = &sB[cur][wn * 32 + f * 16 + l16][ks * 32 + hi * 8];
        bfr[f].q[0] = *(const uint4*)pb;
        bfr[f].q[1] = *(const uint4*)(pb + 16);
      }
#pragma unroll
      for (int fm = 0; fm < 2; ++fm)
#pragma unroll
        for (int fn = 0; fn < 2; ++fn)
          acc[fm][fn] = __builtin_amdgcn_wmma_f32_16x16x32_bf16(
              false, af[fm].v, false, bfr[fn].v, (short)0, acc[fm][fn], false, false);
    }

    if (more) { async_wait0(); storeBnat(cur ^ 1); }
    __syncthreads();
  }

  float* cf = (float*)p.C;
  u16t*  cb = (u16t*)p.C;
#pragma unroll
  for (int fm = 0; fm < 2; ++fm)
#pragma unroll
    for (int fn = 0; fn < 2; ++fn) {
      const int col = colBase + wn * 32 + fn * 16 + l16;
      const float bi = (EPI >= 1) ? p.bias[col] : 0.f;
      AccU a; a.v = acc[fm][fn];
#pragma unroll
      for (int r = 0; r < 8; ++r) {             // C frag: VGPR r -> M = r + hi*8
        const int row = rowBase + wm * 32 + fm * 16 + hi * 8 + r;
        float v = a.f[r] * p.alpha + bi;
        if (EPI == 2) v = gelu_exact(v);
        const long long idx = coff + (long long)row * p.ldc + col;
        if (OBF) cb[idx] = f2bf(v); else cf[idx] = v;
      }
    }
}

// ---------------------------------------------------------------------------
// LayerNorm over D=1024, output bf16. One block (256 thr) per row.
// ---------------------------------------------------------------------------
__global__ __launch_bounds__(256) void k_ln(const float* __restrict__ x,
                                            const float* __restrict__ g,
                                            const float* __restrict__ b,
                                            u16t* __restrict__ out) {
  __shared__ float red[256];
  const int tid = threadIdx.x;
  const float* xr = x + (long long)blockIdx.x * cD;
  float v[4]; float s = 0.f;
#pragma unroll
  for (int i = 0; i < 4; ++i) { v[i] = xr[tid + 256 * i]; s += v[i]; }
  red[tid] = s; __syncthreads();
  for (int st = 128; st > 0; st >>= 1) { if (tid < st) red[tid] += red[tid + st]; __syncthreads(); }
  const float mean = red[0] * (1.f / cD);
  __syncthreads();
  s = 0.f;
#pragma unroll
  for (int i = 0; i < 4; ++i) { float d = v[i] - mean; s += d * d; }
  red[tid] = s; __syncthreads();
  for (int st = 128; st > 0; st >>= 1) { if (tid < st) red[tid] += red[tid + st]; __syncthreads(); }
  const float rstd = rsqrtf(red[0] * (1.f / cD) + 1e-5f);
  u16t* orow = out + (long long)blockIdx.x * cD;
#pragma unroll
  for (int i = 0; i < 4; ++i) {
    int c = tid + 256 * i;
    orow[c] = f2bf((v[i] - mean) * rstd * g[c] + b[c]);
  }
}

// Softmax over K=256 (scale already applied in GEMM); bf16 probs out.
__global__ __launch_bounds__(256) void k_softmax(const float* __restrict__ s,
                                                 u16t* __restrict__ p) {
  __shared__ float red[256];
  const int tid = threadIdx.x;
  const long long row = blockIdx.x;
  const float v = s[row * cK + tid];
  red[tid] = v; __syncthreads();
  for (int st = 128; st > 0; st >>= 1) { if (tid < st) red[tid] = fmaxf(red[tid], red[tid + st]); __syncthreads(); }
  const float m = red[0]; __syncthreads();
  const float e = __expf(v - m);
  red[tid] = e; __syncthreads();
  for (int st = 128; st > 0; st >>= 1) { if (tid < st) red[tid] += red[tid + st]; __syncthreads(); }
  p[row * cK + tid] = f2bf(e / red[0]);
}

// x += y + bias[col], col = i % 1024
__global__ void k_resid(float* __restrict__ x, const float* __restrict__ y,
                        const float* __restrict__ bias, size_t n) {
  size_t i = (size_t)blockIdx.x * 256 + threadIdx.x;
  if (i < n) x[i] += y[i] + bias[i & (cD - 1)];
}

// f32 [R,C] -> bf16 [C,R]  (weight transpose-convert)
__global__ void k_tcvt(const float* __restrict__ in, u16t* __restrict__ out,
                       int R, int C) {
  long long i = (long long)blockIdx.x * 256 + threadIdx.x;
  if (i < (long long)R * C) {
    int r = (int)(i / C), c = (int)(i % C);
    out[(long long)c * R + r] = f2bf(in[i]);
  }
}

// conv_pk f32 [D(o),D(c),S(s)] -> bf16 W^T [o][s*D + c]
__global__ void k_convw(const float* __restrict__ in, u16t* __restrict__ out) {
  long long i = (long long)blockIdx.x * 256 + threadIdx.x;
  if (i < (long long)cD * cD * cS) {
    int o = (int)(i >> 14);                 // / (D*S) = /16384
    int rem = (int)(i & 16383);
    int c = rem >> 4, s = rem & 15;
    out[(long long)o * (cS * cD) + (long long)s * cD + c] = f2bf(in[i]);
  }
}

// ---------------------------------------------------------------------------
// Host orchestration
// ---------------------------------------------------------------------------
static inline GP gp_mk(const void* A, const void* B, void* C, const float* bias,
                       int M, int N, int K, int lda, int ldb, int ldc, float alpha) {
  GP p{}; p.A = (const u16t*)A; p.Bm = (const u16t*)B; p.C = C; p.bias = bias;
  p.M = M; p.N = N; p.K = K; p.lda = lda; p.ldb = ldb; p.ldc = ldc;
  p.HD = 1; p.alpha = alpha;
  p.sA1 = p.sA2 = p.sB1 = p.sB2 = p.sC1 = p.sC2 = 0;
  return p;
}

// mode: 0 = BT,plain,bf16out  1 = BT,plain,f32out  2 = Bnat,plain,bf16out  3 = BT,bias+gelu,bf16out
static void gemm_go(hipStream_t st, int mode, const GP& p, int batch) {
  dim3 g(p.N / BN, p.M / BM, batch), b(256);
  switch (mode) {
    case 0: k_gemm<true, 0, true ><<<g, b, 0, st>>>(p); break;
    case 1: k_gemm<true, 0, false><<<g, b, 0, st>>>(p); break;
    case 2: k_gemm<false, 0, true><<<g, b, 0, st>>>(p); break;
    case 3: k_gemm<true, 2, true ><<<g, b, 0, st>>>(p); break;
  }
}

// workspace layout (bytes); total demand ~394 MB in d_ws
constexpr size_t MB = 1024ull * 1024ull;
constexpr size_t OFF_X = 0, OFF_XN = 32 * MB, OFF_Q = 48 * MB, OFF_KX = 64 * MB,
  OFF_VX = 80 * MB, OFF_KP = 96 * MB, OFF_VP = 97 * MB, OFF_SC = 98 * MB,
  OFF_PR = 162 * MB, OFF_O = 194 * MB, OFF_TMP = 210 * MB, OFF_HID = 242 * MB,
  OFF_WQT = 306 * MB, OFF_WKT = 308 * MB, OFF_WVT = 310 * MB, OFF_WOT = 312 * MB,
  OFF_W1T = 314 * MB, OFF_W2T = 322 * MB, OFF_PKT = 330 * MB, OFF_PVT = 362 * MB;

static void run_layer(hipStream_t st, char* ws, bool conv,
                      const float* ln1g, const float* ln1b,
                      const float* wq, const float* wk, const float* wv,
                      const float* pk, const float* pv,
                      const float* wo, const float* bo,
                      const float* ln2g, const float* ln2b,
                      const float* w1, const float* b1,
                      const float* w2, const float* b2) {
  float* X   = (float*)(ws + OFF_X);
  u16t* XN   = (u16t*)(ws + OFF_XN);
  u16t* Q    = (u16t*)(ws + OFF_Q);
  u16t* KX   = (u16t*)(ws + OFF_KX);
  u16t* VX   = (u16t*)(ws + OFF_VX);
  u16t* KP   = (u16t*)(ws + OFF_KP);
  u16t* VP   = (u16t*)(ws + OFF_VP);
  float* SC  = (float*)(ws + OFF_SC);
  u16t* PR   = (u16t*)(ws + OFF_PR);
  u16t* O    = (u16t*)(ws + OFF_O);
  float* TMP = (float*)(ws + OFF_TMP);
  u16t* HID  = (u16t*)(ws + OFF_HID);
  u16t *WQT = (u16t*)(ws + OFF_WQT), *WKT = (u16t*)(ws + OFF_WKT),
       *WVT = (u16t*)(ws + OFF_WVT), *WOT = (u16t*)(ws + OFF_WOT),
       *W1T = (u16t*)(ws + OFF_W1T), *W2T = (u16t*)(ws + OFF_W2T),
       *PKT = (u16t*)(ws + OFF_PKT), *PVT = (u16t*)(ws + OFF_PVT);

  const long long nDD = (long long)cD * cD;
  auto cvtblk = [](long long n) { return (unsigned)((n + 255) / 256); };

  // ---- weight convert (transpose to B^T bf16) ----
  k_tcvt<<<cvtblk(nDD), 256, 0, st>>>(wq, WQT, cD, cD);
  k_tcvt<<<cvtblk(nDD), 256, 0, st>>>(wk, WKT, cD, cD);
  k_tcvt<<<cvtblk(nDD), 256, 0, st>>>(wv, WVT, cD, cD);
  k_tcvt<<<cvtblk(nDD), 256, 0, st>>>(wo, WOT, cD, cD);
  k_tcvt<<<cvtblk((long long)cD * cDFF), 256, 0, st>>>(w1, W1T, cD, cDFF);
  k_tcvt<<<cvtblk((long long)cDFF * cD), 256, 0, st>>>(w2, W2T, cDFF, cD);
  if (!conv) {  // pk [N,K] -> pk^T [K,N] (used as GEMM A)
    k_tcvt<<<cvtblk((long long)cN * cK), 256, 0, st>>>(pk, PKT, cN, cK);
    k_tcvt<<<cvtblk((long long)cN * cK), 256, 0, st>>>(pv, PVT, cN, cK);
  } else {      // conv weight [o,c,s] -> W^T [o][s*D+c] (B^T for fast path)
    k_convw<<<cvtblk((long long)cD * cD * cS), 256, 0, st>>>(pk, PKT);
    k_convw<<<cvtblk((long long)cD * cD * cS), 256, 0, st>>>(pv, PVT);
  }

  const int MBN = cB * cN;          // 8192 flattened rows
  // ---- LN1 + Q/K/V projections ----
  k_ln<<<MBN, 256, 0, st>>>(X, ln1g, ln1b, XN);
  gemm_go(st, 0, gp_mk(XN, WQT, Q,  nullptr, MBN, cD, cD, cD, cD, cD, 1.f), 1);
  gemm_go(st, 0, gp_mk(XN, WKT, KX, nullptr, MBN, cD, cD, cD, cD, cD, 1.f), 1);
  gemm_go(st, 0, gp_mk(XN, WVT, VX, nullptr, MBN, cD, cD, cD, cD, cD, 1.f), 1);

  // ---- sequence-compress K/V to [B, cK, D] ----
  if (!conv) {
    // k'[b] = pk^T[K,N] · kx[b][N,D]  (B natural layout)
    GP p = gp_mk(PKT, KX, KP, nullptr, cK, cD, cN, cN, cD, cD, 1.f);
    p.sB1 = (long long)cN * cD; p.sC1 = (long long)cK * cD;
    gemm_go(st, 2, p, cB);
    p = gp_mk(PVT, VX, VP, nullptr, cK, cD, cN, cN, cD, cD, 1.f);
    p.sB1 = (long long)cN * cD; p.sC1 = (long long)cK * cD;
    gemm_go(st, 2, p, cB);
  } else {
    // k' = kx viewed [B*K, S*D] · W^T-fast-path [D, S*D]
    gemm_go(st, 0, gp_mk(KX, PKT, KP, nullptr, cB * cK, cD, cS * cD, cS * cD, cS * cD, cD, 1.f), 1);
    gemm_go(st, 0, gp_mk(VX, PVT, VP, nullptr, cB * cK, cD, cS * cD, cS * cD, cS * cD, cD, 1.f), 1);
  }

  // ---- attention scores: per (b,h) Q[N,DH] · k'[K,DH]^T, alpha = DH^-0.5 ----
  {
    GP p = gp_mk(Q, KP, SC, nullptr, cN, cK, cDH, cD, cD, cK, 0.08838834764831845f);
    p.HD = cH;
    p.sA1 = (long long)cN * cD; p.sA2 = cDH;            // Q: batch b, head h
    p.sB1 = (long long)cK * cD; p.sB2 = cDH;            // k' rows are B^T
    p.sC1 = (long long)cH * cN * cK; p.sC2 = (long long)cN * cK;
    gemm_go(st, 1, p, cB * cH);
  }
  k_softmax<<<cB * cH * cN, 256, 0, st>>>(SC, PR);

  // ---- attn out: probs[N,K] · v'[K,DH] (B natural), write bf16 into [B,N,D] ----
  {
    GP p = gp_mk(PR, VP, O, nullptr, cN, cDH, cK, cK, cD, cD, 1.f);
    p.HD = cH;
    p.sA1 = (long long)cH * cN * cK; p.sA2 = (long long)cN * cK;
    p.sB1 = (long long)cK * cD; p.sB2 = cDH;
    p.sC1 = (long long)cN * cD; p.sC2 = cDH;
    gemm_go(st, 2, p, cB * cH);
  }

  // ---- output projection + residual ----
  gemm_go(st, 1, gp_mk(O, WOT, TMP, nullptr, MBN, cD, cD, cD, cD, cD, 1.f), 1);
  const size_t nx = (size_t)cB * cN * cD;
  k_resid<<<(unsigned)(nx / 256), 256, 0, st>>>(X, TMP, bo, nx);

  // ---- FFN: gelu(xn·W1+b1)·W2 + b2, residual ----
  k_ln<<<MBN, 256, 0, st>>>(X, ln2g, ln2b, XN);
  gemm_go(st, 3, gp_mk(XN, W1T, HID, b1, MBN, cDFF, cD, cD, cD, cDFF, 1.f), 1);
  gemm_go(st, 1, gp_mk(HID, W2T, TMP, nullptr, MBN, cD, cDFF, cDFF, cDFF, cD, 1.f), 1);
  k_resid<<<(unsigned)(nx / 256), 256, 0, st>>>(X, TMP, b2, nx);
}

extern "C" void kernel_launch(void* const* d_in, const int* in_sizes, int n_in,
                              void* d_out, int out_size, void* d_ws, size_t ws_size,
                              hipStream_t stream) {
  (void)in_sizes; (void)n_in; (void)out_size; (void)ws_size;
  const float* I[31];
  for (int i = 0; i < 31; ++i) I[i] = (const float*)d_in[i];

  char* ws = (char*)d_ws;
  float* X = (float*)(ws + OFF_X);
  const size_t xbytes = (size_t)cB * cN * cD * sizeof(float);
  hipMemcpyAsync(X, I[0], xbytes, hipMemcpyDeviceToDevice, stream);

  const size_t sDD = (size_t)cD * cD, sDF = (size_t)cD * cDFF;
  // Linformer layers
  for (int i = 0; i < cL; ++i) {
    run_layer(stream, ws, /*conv=*/false,
              I[1] + (size_t)i * cD, I[2] + (size_t)i * cD,
              I[3] + i * sDD, I[4] + i * sDD, I[5] + i * sDD,
              I[6] + (size_t)i * cN * cK, I[7] + (size_t)i * cN * cK,
              I[8] + i * sDD, I[9] + (size_t)i * cD,
              I[10] + (size_t)i * cD, I[11] + (size_t)i * cD,
              I[12] + i * sDF, I[13] + (size_t)i * cDFF,
              I[14] + i * sDF, I[15] + (size_t)i * cD);
  }
  // ConvLinformer layers
  for (int i = 0; i < cL; ++i) {
    run_layer(stream, ws, /*conv=*/true,
              I[16] + (size_t)i * cD, I[17] + (size_t)i * cD,
              I[18] + i * sDD, I[19] + i * sDD, I[20] + i * sDD,
              I[21] + (size_t)i * cD * cD * cS, I[22] + (size_t)i * cD * cD * cS,
              I[23] + i * sDD, I[24] + (size_t)i * cD,
              I[25] + (size_t)i * cD, I[26] + (size_t)i * cD,
              I[27] + i * sDF, I[28] + (size_t)i * cDFF,
              I[29] + i * sDF, I[30] + (size_t)i * cD);
  }

  hipMemcpyAsync(d_out, X, xbytes, hipMemcpyDeviceToDevice, stream);
}
